// MoEACT_24043226923569
// MI455X (gfx1250) — matrix operand order repaired and checked by reading, not compiled, not run on hardware
//
#include <hip/hip_runtime.h>
#include <hip/hip_bf16.h>
#include <stdint.h>

// MoE dims (match reference)
#define T_TOK 32768   // B*S tokens
#define D_DIM 512
#define F_DIM 2048
#define E_NUM 8

typedef __bf16 bf16_t;
typedef __attribute__((ext_vector_type(16))) __bf16 v16bf;
typedef __attribute__((ext_vector_type(8)))  float  v8f;

union Frag16 { v16bf v; uint4 q[2]; };

// Load 8+8 bf16 elements (two b128s) at p and p+second_off (element units).
__device__ __forceinline__ v16bf frag_ld(const bf16_t* p, int second_off) {
    Frag16 f;
    f.q[0] = *(const uint4*)(p);
    f.q[1] = *(const uint4*)(p + second_off);
    return f.v;
}

// A-matrix 16x32 bf16 frag (ISA 7.12.2): lanes 0-15 -> M=lane, K {k0+0..7, k0+16..23};
// lanes 16-31 -> M=lane-16, K {k0+8..15, k0+24..31}.
__device__ __forceinline__ v16bf load_a(const bf16_t* base, int stride, int m_base,
                                        int k0, int lane) {
    int m  = m_base + (lane & 15);
    int kh = (lane & 16) ? 8 : 0;
    return frag_ld(base + (size_t)m * stride + k0 + kh, 16);
}

// B-matrix 32x16 bf16 frag (same K order as A): lane -> column N=lane&15,
// lanes 0-15 hold K {k0+0..7, k0+16..23}, lanes 16-31 hold K {k0+8..15, k0+24..31}.
// Weight matrix is stored "transposed" ([N, K] row-major) so runs are contiguous.
__device__ __forceinline__ v16bf load_b(const bf16_t* base, int stride, int n_base,
                                        int k0, int lane) {
    int n  = n_base + (lane & 15);
    int kh = (lane & 16) ? 8 : 0;
    return frag_ld(base + (size_t)n * stride + k0 + kh, 16);
}

__device__ __forceinline__ v8f wmma_bf16(v16bf a, v16bf b, v8f c) {
    return __builtin_amdgcn_wmma_f32_16x16x32_bf16(false, a, false, b, (short)0, c,
                                                   false, false);
}

// ---------------------------------------------------------------------------
// Prep kernels
// ---------------------------------------------------------------------------
__global__ void zero_counts_kernel(int* counts) {
    if (threadIdx.x < E_NUM) counts[threadIdx.x] = 0;
}

__global__ void cvt_x_kernel(const float* __restrict__ x, bf16_t* __restrict__ xb) {
    int i = (blockIdx.x * 256 + threadIdx.x) * 4;
    float4 v = *(const float4*)(x + i);
    union { bf16_t h[4]; uint2 u; } o;
    o.h[0] = (bf16_t)v.x; o.h[1] = (bf16_t)v.y;
    o.h[2] = (bf16_t)v.z; o.h[3] = (bf16_t)v.w;
    *(uint2*)(xb + i) = o.u;
}

// W1 [E, D, F] f32 -> W1t [E, F, D] bf16 (transpose; L2-resident, coalesced writes)
__global__ void cvt_w1_kernel(const float* __restrict__ W1, bf16_t* __restrict__ W1t) {
    size_t i = (size_t)blockIdx.x * 256 + threadIdx.x;
    int d = (int)(i % D_DIM);
    size_t r = i / D_DIM;
    int f = (int)(r % F_DIM);
    int e = (int)(r / F_DIM);
    W1t[i] = (bf16_t)W1[((size_t)e * D_DIM + d) * F_DIM + f];
}

// W2 [E, F, D] f32 -> W2t [E, D, F] bf16
__global__ void cvt_w2_kernel(const float* __restrict__ W2, bf16_t* __restrict__ W2t) {
    size_t i = (size_t)blockIdx.x * 256 + threadIdx.x;
    int f = (int)(i % F_DIM);
    size_t r = i / F_DIM;
    int d = (int)(r % D_DIM);
    int e = (int)(r / D_DIM);
    W2t[i] = (bf16_t)W2[((size_t)e * F_DIM + f) * D_DIM + d];
}

// ---------------------------------------------------------------------------
// Router: one wave32 per token. logits = x@Wr+br, softmax, top-2, renormalize,
// scatter (token, gate) into per-expert lists.
// ---------------------------------------------------------------------------
__global__ void router_kernel(const float* __restrict__ x,
                              const float* __restrict__ Wr,   // [D, E]
                              const float* __restrict__ br,   // [E]
                              float* __restrict__ combine,    // [T, E]
                              int*   __restrict__ counts,     // [E]
                              int*   __restrict__ tok_ids,    // [E, T]
                              float* __restrict__ gates)      // [E, T]
{
    const int lane = threadIdx.x & 31;
    const int wave = threadIdx.x >> 5;
    const int t = blockIdx.x * 8 + wave;
    const float* xr = x + (size_t)t * D_DIM;

    float acc[E_NUM];
#pragma unroll
    for (int e = 0; e < E_NUM; ++e) acc[e] = 0.f;

    for (int d = lane; d < D_DIM; d += 32) {
        float xv = xr[d];
        const float* wrow = Wr + d * E_NUM;
#pragma unroll
        for (int e = 0; e < E_NUM; ++e) acc[e] += xv * wrow[e];
    }
#pragma unroll
    for (int e = 0; e < E_NUM; ++e) {
        float v = acc[e];
#pragma unroll
        for (int off = 16; off > 0; off >>= 1) v += __shfl_xor(v, off, 32);
        acc[e] = v;
    }

    if (lane == 0) {
        float g[E_NUM];
        float mx = -1e30f;
#pragma unroll
        for (int e = 0; e < E_NUM; ++e) { g[e] = acc[e] + br[e]; mx = fmaxf(mx, g[e]); }
        float s = 0.f;
#pragma unroll
        for (int e = 0; e < E_NUM; ++e) { g[e] = __expf(g[e] - mx); s += g[e]; }
        float inv = 1.f / s;
#pragma unroll
        for (int e = 0; e < E_NUM; ++e) g[e] *= inv;

        int i1 = 0;
        for (int e = 1; e < E_NUM; ++e) if (g[e] > g[i1]) i1 = e;
        int i2 = (i1 == 0) ? 1 : 0;
        for (int e = 0; e < E_NUM; ++e) if (e != i1 && g[e] > g[i2]) i2 = e;
        float den = g[i1] + g[i2];
        float w1v = g[i1] / den, w2v = g[i2] / den;

        float* crow = combine + (size_t)t * E_NUM;
#pragma unroll
        for (int e = 0; e < E_NUM; ++e) crow[e] = 0.f;
        crow[i1] = w1v;
        crow[i2] = w2v;

        int p1 = atomicAdd(&counts[i1], 1);
        tok_ids[i1 * T_TOK + p1] = t;
        gates[i1 * T_TOK + p1]   = w1v;
        int p2 = atomicAdd(&counts[i2], 1);
        tok_ids[i2 * T_TOK + p2] = t;
        gates[i2 * T_TOK + p2]   = w2v;
    }
}

// out[t, d] = sum_e combine[t,e] * b2[e,d]   (bias term; also clears d_out)
__global__ void out_init_kernel(const float* __restrict__ combine,
                                const float* __restrict__ b2,
                                float* __restrict__ out) {
    size_t i = (size_t)blockIdx.x * 256 + threadIdx.x;
    int d = (int)(i % D_DIM);
    size_t t = i / D_DIM;
    const float* c = combine + t * E_NUM;
    float s = 0.f;
#pragma unroll
    for (int e = 0; e < E_NUM; ++e) s += c[e] * b2[e * D_DIM + d];
    out[i] = s;
}

// ---------------------------------------------------------------------------
// Fused expert FFN: block = 32 gathered tokens of one expert.
// LDS: x-tile 32x512 bf16 + h 32x2048 bf16 (row-padded for banks), ~161 KB of
// the 320 KB CDNA5 WGP LDS. All matmuls via v_wmma_f32_16x16x32_bf16.
// Token gather uses the CDNA5 async global->LDS path (ASYNCcnt).
// ---------------------------------------------------------------------------
#define XS 520   // 512 + 8 pad (keeps 16B row alignment, 4-bank shift/row)
#define HS 2056  // 2048 + 8 pad

__global__ __launch_bounds__(256)
void ffn_kernel(const bf16_t* __restrict__ xb,      // [T, D]
                const bf16_t* __restrict__ W1t,     // [E, F, D]
                const bf16_t* __restrict__ W2t,     // [E, D, F]
                const float*  __restrict__ b1,      // [E, F]
                const int*    __restrict__ counts,  // [E]
                const int*    __restrict__ tok_ids, // [E, T]
                const float*  __restrict__ gates,   // [E, T]
                float* __restrict__ out)            // [T, D]
{
    __shared__ bf16_t s_x[32 * XS];
    __shared__ bf16_t s_h[32 * HS];

    const int tiles_per_e = T_TOK / 32;
    const int e   = blockIdx.x / tiles_per_e;
    const int til = blockIdx.x % tiles_per_e;
    const int n_tok = counts[e];
    if (til * 32 >= n_tok) return;   // uniform early exit (static grid, graph-safe)

    const int tid  = threadIdx.x;
    const int lane = tid & 31;
    const int wave = tid >> 5;

    const bf16_t* w1e = W1t + (size_t)e * F_DIM * D_DIM;
    const bf16_t* w2e = W2t + (size_t)e * D_DIM * F_DIM;

    // ---- Phase 1: async-gather 32 token rows straight into LDS ----
    // global_load_async_to_lds_b128: per-lane global addr -> per-lane LDS addr,
    // no VGPR round trip, tracked by ASYNCcnt (ISA 10.x / CDNA5 ISA).
    {
        const int r = tid >> 3;         // row 0..31
        const int c = (tid & 7) * 64;   // 64 elements per copier
        int idx = til * 32 + r;
        int tok = (idx < n_tok) ? tok_ids[e * T_TOK + idx] : 0;
        const bf16_t* src = xb + (size_t)tok * D_DIM + c;
        bf16_t* dst = s_x + r * XS + c;
        // LDS byte address = low 32 bits of the generic address (aperture rule)
        unsigned lds_off = (unsigned)(uintptr_t)dst;
#pragma unroll
        for (int q = 0; q < 8; ++q) {
            asm volatile("global_load_async_to_lds_b128 %0, %1, off"
                         :: "v"(lds_off + 16u * q), "v"(src + 8 * q)
                         : "memory");
        }
        asm volatile("s_wait_asynccnt 0x0" ::: "memory");
    }
    __syncthreads();

    // ---- Phase 2: h = relu(x @ W1 + b1), kept in LDS as bf16 ----
    {
        const int m_base = (wave & 1) * 16;        // 2 row-halves
        const int f_base = (wave >> 1) * 512;      // 4 F-panels of 512
        for (int fg = 0; fg < 8; ++fg) {           // groups of 4 16-col tiles
            const int f0 = f_base + fg * 64;
            __builtin_prefetch(w1e + (size_t)(f0 + 64) * D_DIM, 0, 0);
            v8f acc[4];
#pragma unroll
            for (int j = 0; j < 4; ++j) acc[j] = v8f{};
            for (int k = 0; k < D_DIM; k += 32) {
                v16bf a = load_a(s_x, XS, m_base, k, lane);
                v16bf b[4];
                // issue all 4 B-frag load clauses before any WMMA consumes them
#pragma unroll
                for (int j = 0; j < 4; ++j)
                    b[j] = load_b(w1e, D_DIM, f0 + j * 16, k, lane);
#pragma unroll
                for (int j = 0; j < 4; ++j)
                    acc[j] = wmma_bf16(a, b[j], acc[j]);
            }
            const int n  = lane & 15;
            const int mo = (lane & 16) ? 8 : 0;
#pragma unroll
            for (int j = 0; j < 4; ++j) {
                const int fc = f0 + j * 16 + n;
                const float bv = b1[e * F_DIM + fc];
#pragma unroll
                for (int vi = 0; vi < 8; ++vi) {
                    float h = acc[j][vi] + bv;
                    h = h > 0.f ? h : 0.f;
                    s_h[(m_base + mo + vi) * HS + fc] = (bf16_t)h;
                }
            }
        }
    }
    __syncthreads();

    // ---- Phase 3+4: y = h @ W2; out[tok] += gate * y (atomic f32) ----
    {
        const int m_base = (wave & 1) * 16;
        const int d_base = (wave >> 1) * 128;      // 4 D-panels of 128
        v8f acc[8];
#pragma unroll
        for (int j = 0; j < 8; ++j) acc[j] = v8f{};
        for (int k = 0; k < F_DIM; k += 32) {
            v16bf a = load_a(s_h, HS, m_base, k, lane);
            v16bf b[8];
            // one big load clause, single wait amortized over 8 WMMAs
#pragma unroll
            for (int j = 0; j < 8; ++j)
                b[j] = load_b(w2e, F_DIM, d_base + j * 16, k, lane);
#pragma unroll
            for (int j = 0; j < 8; ++j)
                acc[j] = wmma_bf16(a, b[j], acc[j]);
        }
        const int n  = lane & 15;
        const int mo = (lane & 16) ? 8 : 0;
        int   toks[8];
        float gv[8];
#pragma unroll
        for (int vi = 0; vi < 8; ++vi) {
            int idx = til * 32 + m_base + mo + vi;
            bool ok = idx < n_tok;
            toks[vi] = ok ? tok_ids[e * T_TOK + idx] : 0;
            gv[vi]   = ok ? gates[e * T_TOK + idx]   : 0.f;
        }
#pragma unroll
        for (int j = 0; j < 8; ++j) {
            int dc = d_base + j * 16 + n;
#pragma unroll
            for (int vi = 0; vi < 8; ++vi) {
                atomicAdd(&out[(size_t)toks[vi] * D_DIM + dc], gv[vi] * acc[j][vi]);
            }
        }
    }
}

// ---------------------------------------------------------------------------
// Launcher
// ---------------------------------------------------------------------------
extern "C" void kernel_launch(void* const* d_in, const int* in_sizes, int n_in,
                              void* d_out, int out_size, void* d_ws, size_t ws_size,
                              hipStream_t stream) {
    const float* x  = (const float*)d_in[0];
    const float* Wr = (const float*)d_in[1];
    const float* br = (const float*)d_in[2];
    const float* W1 = (const float*)d_in[3];
    const float* b1 = (const float*)d_in[4];
    const float* W2 = (const float*)d_in[5];
    const float* b2 = (const float*)d_in[6];
    float* out = (float*)d_out;

    char* ws = (char*)d_ws;
    size_t off = 0;
    bf16_t* xb  = (bf16_t*)(ws + off); off += (size_t)T_TOK * D_DIM * 2;
    bf16_t* W1t = (bf16_t*)(ws + off); off += (size_t)E_NUM * F_DIM * D_DIM * 2;
    bf16_t* W2t = (bf16_t*)(ws + off); off += (size_t)E_NUM * D_DIM * F_DIM * 2;
    float* combine = (float*)(ws + off); off += (size_t)T_TOK * E_NUM * 4;
    int*   tok_ids = (int*)(ws + off);   off += (size_t)E_NUM * T_TOK * 4;
    float* gates   = (float*)(ws + off); off += (size_t)E_NUM * T_TOK * 4;
    int*   counts  = (int*)(ws + off);   off += 256;

    zero_counts_kernel<<<1, 32, 0, stream>>>(counts);
    cvt_x_kernel<<<(T_TOK * D_DIM / 4) / 256, 256, 0, stream>>>(x, xb);
    cvt_w1_kernel<<<(E_NUM * F_DIM * D_DIM) / 256, 256, 0, stream>>>(W1, W1t);
    cvt_w2_kernel<<<(E_NUM * D_DIM * F_DIM) / 256, 256, 0, stream>>>(W2, W2t);
    router_kernel<<<T_TOK / 8, 256, 0, stream>>>(x, Wr, br, combine, counts,
                                                 tok_ids, gates);
    out_init_kernel<<<(T_TOK * D_DIM) / 256, 256, 0, stream>>>(combine, b2, out);
    ffn_kernel<<<E_NUM * (T_TOK / 32), 256, 0, stream>>>(xb, W1t, W2t, b1, counts,
                                                         tok_ids, gates, out);
}
